// DynamicFFNLayer_37417755083479
// MI455X (gfx1250) — matrix operand order repaired and checked by reading, not compiled, not run on hardware
//
#include <hip/hip_runtime.h>
#include <cstdint>
#include <cstddef>

// ---------------------------------------------------------------------------
// DynamicFFNLayer for MI455X (gfx1250, wave32, WMMA).
// - All large matmuls on v_wmma_f32_16x16x32_bf16
// - Tile staging via CDNA5 async path: global_load_async_to_lds_b128 with
//   double-buffered LDS, pipelined with s_wait_asynccnt (ASYNCcnt machinery)
// - Fused epilogues: gelu in GEMM; rho2 (128->1) dot fused into rho1 GEMM
// - Top-k = per-token 8-bit radix select (index order irrelevant to the math)
// ---------------------------------------------------------------------------

typedef __bf16 bf16_t;
typedef __attribute__((ext_vector_type(16))) __bf16 v16bf;
typedef __attribute__((ext_vector_type(8)))  float  v8f;

#define T_TOK 4096
#define DM    1024
#define DF    4096
#define KSEL  256
#define DH    64
#define DH2   32

__device__ __forceinline__ float gelu_f(float x) {
  return 0.5f * x * (1.0f + erff(x * 0.70710678118654752f));
}
__device__ __forceinline__ int imin(int a, int b) { return a < b ? a : b; }

// CDNA5 async LDS fill: writes 16B straight to LDS, tracked by ASYNCcnt.
__device__ __forceinline__ void async_b128(uint32_t lds_addr, uint64_t gaddr) {
  asm volatile("global_load_async_to_lds_b128 %0, %1, off"
               :: "v"(lds_addr), "v"(gaddr) : "memory");
}
__device__ __forceinline__ uint32_t lds_off(const void* p) {
  return (uint32_t)(uintptr_t)p;   // low 32 bits of flat LDS addr == LDS offset
}

// ------------------------- generic WMMA GEMM: C = A @ B^T -------------------
// A: [M,Kd] bf16 row-major, B: [N,Kd] bf16 row-major (weights are [out,in]).
// Block tile 64x128, 8 waves (2x4), each wave 32x32 (2x2 WMMA 16x16x32 tiles).
// Double-buffered LDS filled by async loads; OOB rows are clamped (their
// results only feed outputs discarded by the m<M / n<N store guards).
template<bool GELU, bool OUT_BF16>
__global__ __launch_bounds__(256) void wmma_gemm_kernel(
    const bf16_t* __restrict__ A, const bf16_t* __restrict__ B,
    void* __restrict__ Cout, const float* __restrict__ bias,
    int M, int N, int Kd)
{
  __shared__ __align__(16) bf16_t As[2][64][40];    // +8 pad vs bank conflicts
  __shared__ __align__(16) bf16_t Bs[2][128][40];
  const int tid  = threadIdx.x;
  const int lane = tid & 31;
  const int wave = tid >> 5;
  const int wr = wave >> 2;        // 0..1  (row group of 32)
  const int wc = wave & 3;         // 0..3  (col group of 32)
  const int hh  = lane >> 4;       // lane half
  const int l15 = lane & 15;
  const int bm = blockIdx.x * 64;
  const int bn = blockIdx.y * 128;

  // per-thread tile-fill assignment (k-invariant)
  const int ra = tid >> 2, ca = (tid & 3) * 8;          // A: 64 rows x 4 segs
  const int rb = tid >> 1, cb = (tid & 1) * 16;         // B: 128 rows x 2 segs
  const bf16_t* Ag = A + (size_t)imin(bm + ra, M - 1) * Kd + ca;
  const bf16_t* Bg = B + (size_t)imin(bn + rb, N - 1) * Kd + cb;

  auto issue = [&](int buf, int k0) {
    async_b128(lds_off(&As[buf][ra][ca]),    (uint64_t)(uintptr_t)(Ag + k0));
    async_b128(lds_off(&Bs[buf][rb][cb]),    (uint64_t)(uintptr_t)(Bg + k0));
    async_b128(lds_off(&Bs[buf][rb][cb + 8]),(uint64_t)(uintptr_t)(Bg + k0 + 8));
  };

  v8f acc[2][2];
  #pragma unroll
  for (int i = 0; i < 2; ++i)
    #pragma unroll
    for (int j = 0; j < 2; ++j)
      acc[i][j] = (v8f){0.f,0.f,0.f,0.f,0.f,0.f,0.f,0.f};

  issue(0, 0);
  int p = 0;
  for (int k0 = 0; k0 < Kd; k0 += 32) {
    const bool more = (k0 + 32) < Kd;
    if (more) issue(p ^ 1, k0 + 32);
    if (more) asm volatile("s_wait_asynccnt 0x3" ::: "memory");
    else      asm volatile("s_wait_asynccnt 0x0" ::: "memory");
    __syncthreads();

    // Fragments per ISA VGPR layouts:
    //  A 16x32 bf16: lane -> M=l15; e<8 -> K=e, e>=8 -> K=e+8; +8 for hh=1
    //  B 32x16 bf16: lane -> N=l15; e<8 -> K=e, e>=8 -> K=e+8; +8 for hh=1
    v16bf afr[2], bfr[2];
    #pragma unroll
    for (int s = 0; s < 2; ++s) {
      const int row = wr * 32 + s * 16 + l15;
      #pragma unroll
      for (int e = 0; e < 16; ++e)
        afr[s][e] = As[p][row][(e < 8 ? e : e + 8) + (hh ? 8 : 0)];
      const int col = wc * 32 + s * 16 + l15;
      #pragma unroll
      for (int e = 0; e < 16; ++e)
        bfr[s][e] = Bs[p][col][(e < 8 ? e : e + 8) + (hh ? 8 : 0)];
    }
    #pragma unroll
    for (int i = 0; i < 2; ++i)
      #pragma unroll
      for (int j = 0; j < 2; ++j)
        acc[i][j] = __builtin_amdgcn_wmma_f32_16x16x32_bf16(
            false, afr[i], false, bfr[j], (short)0, acc[i][j], false, false);
    __syncthreads();
    p ^= 1;
  }

  // C/D layout: lane -> N=l15; VGPR r -> M = r + 8*hh
  #pragma unroll
  for (int i = 0; i < 2; ++i)
    #pragma unroll
    for (int j = 0; j < 2; ++j)
      #pragma unroll
      for (int r = 0; r < 8; ++r) {
        const int m = bm + wr * 32 + i * 16 + r + (hh ? 8 : 0);
        const int n = bn + wc * 32 + j * 16 + l15;
        if (m < M && n < N) {
          float v = acc[i][j][r];
          if (bias) v += bias[n];
          if (GELU) v = gelu_f(v);
          if (OUT_BF16) ((bf16_t*)Cout)[(size_t)m * N + n] = (bf16_t)v;
          else          ((float*)Cout)[(size_t)m * N + n]  = v;
        }
      }
}

// ----------- rho GEMM: man[m] = rho2_b + sum_n gelu(E@rho1^T + b)[m,n]*rho2_w[n]
// M = T*K rows, N=128, Kd=64. One block covers 64 rows x all 128 cols, so the
// rho2 dot reduces entirely inside the block (LDS float atomics), no 256MB temp.
__global__ __launch_bounds__(256) void wmma_rho_kernel(
    const bf16_t* __restrict__ A, const bf16_t* __restrict__ B,
    const float* __restrict__ bias, const float* __restrict__ rho2_w,
    const float* __restrict__ rho2_b, float* __restrict__ man, int M)
{
  const int N = 128, Kd = 64;
  __shared__ __align__(16) bf16_t As[2][64][40];
  __shared__ __align__(16) bf16_t Bs[2][128][40];
  __shared__ float man_s[64];
  const int tid  = threadIdx.x;
  const int lane = tid & 31;
  const int wave = tid >> 5;
  const int wr = wave >> 2, wc = wave & 3;
  const int hh = lane >> 4, l15 = lane & 15;
  const int bm = blockIdx.x * 64;

  if (tid < 64) man_s[tid] = 0.0f;

  const int ra = tid >> 2, ca = (tid & 3) * 8;
  const int rb = tid >> 1, cb = (tid & 1) * 16;
  const bf16_t* Ag = A + (size_t)imin(bm + ra, M - 1) * Kd + ca;
  const bf16_t* Bg = B + (size_t)rb * Kd + cb;

  auto issue = [&](int buf, int k0) {
    async_b128(lds_off(&As[buf][ra][ca]),     (uint64_t)(uintptr_t)(Ag + k0));
    async_b128(lds_off(&Bs[buf][rb][cb]),     (uint64_t)(uintptr_t)(Bg + k0));
    async_b128(lds_off(&Bs[buf][rb][cb + 8]), (uint64_t)(uintptr_t)(Bg + k0 + 8));
  };

  v8f acc[2][2];
  #pragma unroll
  for (int i = 0; i < 2; ++i)
    #pragma unroll
    for (int j = 0; j < 2; ++j)
      acc[i][j] = (v8f){0.f,0.f,0.f,0.f,0.f,0.f,0.f,0.f};

  issue(0, 0);
  int p = 0;
  for (int k0 = 0; k0 < Kd; k0 += 32) {
    const bool more = (k0 + 32) < Kd;
    if (more) issue(p ^ 1, k0 + 32);
    if (more) asm volatile("s_wait_asynccnt 0x3" ::: "memory");
    else      asm volatile("s_wait_asynccnt 0x0" ::: "memory");
    __syncthreads();

    v16bf afr[2], bfr[2];
    #pragma unroll
    for (int s = 0; s < 2; ++s) {
      const int row = wr * 32 + s * 16 + l15;
      #pragma unroll
      for (int e = 0; e < 16; ++e)
        afr[s][e] = As[p][row][(e < 8 ? e : e + 8) + (hh ? 8 : 0)];
      const int col = wc * 32 + s * 16 + l15;
      #pragma unroll
      for (int e = 0; e < 16; ++e)
        bfr[s][e] = Bs[p][col][(e < 8 ? e : e + 8) + (hh ? 8 : 0)];
    }
    #pragma unroll
    for (int i = 0; i < 2; ++i)
      #pragma unroll
      for (int j = 0; j < 2; ++j)
        acc[i][j] = __builtin_amdgcn_wmma_f32_16x16x32_bf16(
            false, afr[i], false, bfr[j], (short)0, acc[i][j], false, false);
    __syncthreads();
    p ^= 1;
  }

  // Fused epilogue: gelu + bias, multiply by rho2_w[n], reduce over n into rows.
  #pragma unroll
  for (int i = 0; i < 2; ++i)
    #pragma unroll
    for (int j = 0; j < 2; ++j)
      #pragma unroll
      for (int r = 0; r < 8; ++r) {
        const int mloc = wr * 32 + i * 16 + r + (hh ? 8 : 0);
        const int n    = wc * 32 + j * 16 + l15;
        const float v  = gelu_f(acc[i][j][r] + bias[n]);
        atomicAdd(&man_s[mloc], v * rho2_w[n]);
      }
  __syncthreads();
  if (tid < 64) {
    const int m = bm + tid;
    if (m < M) man[m] = man_s[tid] + rho2_b[0];
  }
}

// ------------------------- LayerNorm(x) -> bf16, plus raw x -> bf16 ---------
__global__ __launch_bounds__(256) void lnx_kernel(
    const float* __restrict__ x, const float* __restrict__ w,
    const float* __restrict__ b, bf16_t* __restrict__ xn, bf16_t* __restrict__ xb)
{
  __shared__ float red[256];
  __shared__ float s_mu, s_rstd;
  const int t = blockIdx.x, tid = threadIdx.x;
  const float* row = x + (size_t)t * DM;
  float v[4]; float s = 0.f;
  #pragma unroll
  for (int j = 0; j < 4; ++j) { v[j] = row[tid * 4 + j]; s += v[j]; }
  red[tid] = s; __syncthreads();
  for (int st = 128; st > 0; st >>= 1) { if (tid < st) red[tid] += red[tid + st]; __syncthreads(); }
  if (tid == 0) s_mu = red[0] * (1.0f / DM);
  __syncthreads();
  const float mu = s_mu;
  float sq = 0.f;
  #pragma unroll
  for (int j = 0; j < 4; ++j) { const float d = v[j] - mu; sq += d * d; }
  red[tid] = sq; __syncthreads();
  for (int st = 128; st > 0; st >>= 1) { if (tid < st) red[tid] += red[tid + st]; __syncthreads(); }
  if (tid == 0) s_rstd = rsqrtf(red[0] * (1.0f / DM) + 1e-5f);
  __syncthreads();
  const float rstd = s_rstd;
  #pragma unroll
  for (int j = 0; j < 4; ++j) {
    const int i = tid * 4 + j;
    const float o = (v[j] - mu) * rstd * w[i] + b[i];
    xn[(size_t)t * DM + i] = (bf16_t)o;
    xb[(size_t)t * DM + i] = (bf16_t)v[j];
  }
}

// ------------------------- per-token top-256 radix select ------------------
__global__ __launch_bounds__(256) void topk_kernel(
    const float* __restrict__ scores, int* __restrict__ topidx)
{
  __shared__ unsigned hist[256];
  __shared__ unsigned s_prefix, s_need;
  __shared__ int s_cnt, s_eq;
  const int t = blockIdx.x, tid = threadIdx.x;
  const float* s = scores + (size_t)t * DF;

  unsigned keys[16];
  #pragma unroll
  for (int j = 0; j < 16; ++j) {
    const unsigned u = __float_as_uint(s[tid + j * 256]);
    keys[j] = (u & 0x80000000u) ? ~u : (u | 0x80000000u);  // order-preserving
  }
  if (tid == 0) { s_prefix = 0u; s_need = KSEL; }
  __syncthreads();

  for (int pass = 0; pass < 4; ++pass) {
    const int shift = 24 - 8 * pass;
    hist[tid] = 0u; __syncthreads();
    const unsigned pfx   = s_prefix;
    const unsigned pmask = (pass == 0) ? 0u : (0xFFFFFFFFu << (shift + 8));
    #pragma unroll
    for (int j = 0; j < 16; ++j)
      if ((keys[j] & pmask) == pfx)
        atomicAdd(&hist[(keys[j] >> shift) & 255u], 1u);
    __syncthreads();
    if (tid == 0) {
      unsigned need = s_need, cum = 0; int bsel = 0;
      for (int b = 255; b >= 0; --b) {
        if (cum + hist[b] >= need) { bsel = b; break; }
        cum += hist[b];
      }
      s_prefix = pfx | ((unsigned)bsel << shift);
      s_need   = need - cum;
    }
    __syncthreads();
  }

  const unsigned thr = s_prefix;
  const unsigned need_eq = s_need;
  if (tid == 0) { s_cnt = 0; s_eq = 0; }
  __syncthreads();
  #pragma unroll
  for (int j = 0; j < 16; ++j) {
    const unsigned u = keys[j];
    const int i = tid + j * 256;
    if (u > thr) {
      const int p = atomicAdd(&s_cnt, 1);
      topidx[(size_t)t * KSEL + p] = i;
    } else if (u == thr) {
      const int e = atomicAdd(&s_eq, 1);
      if (e < (int)need_eq) {
        const int p = atomicAdd(&s_cnt, 1);
        topidx[(size_t)t * KSEL + p] = i;
      }
    }
  }
}

// ----------- DeepSets stage 1: gather z, gelu, phi1, LN, gelu -> bf16 [1M,32]
__global__ __launch_bounds__(256) void ds_phi1_kernel(
    const float* __restrict__ z, const int* __restrict__ topidx,
    const float* __restrict__ phi1_w, const float* __restrict__ phi1_b,
    const float* __restrict__ ln1w, const float* __restrict__ ln1b,
    bf16_t* __restrict__ A_ds)
{
  const size_t i = (size_t)blockIdx.x * 256 + threadIdx.x;  // 0..T*K-1
  const int t = (int)(i >> 8);
  const int idx = topidx[i];
  const float a = gelu_f(z[(size_t)t * DF + idx]);
  float v[DH2]; float sum = 0.f;
  #pragma unroll
  for (int j = 0; j < DH2; ++j) { v[j] = a * phi1_w[j] + phi1_b[j]; sum += v[j]; }
  const float mu = sum * (1.0f / DH2);
  float sq = 0.f;
  #pragma unroll
  for (int j = 0; j < DH2; ++j) { const float d = v[j] - mu; sq += d * d; }
  const float rstd = rsqrtf(sq * (1.0f / DH2) + 1e-5f);
  #pragma unroll
  for (int j = 0; j < DH2; ++j)
    A_ds[i * DH2 + j] = (bf16_t)gelu_f((v[j] - mu) * rstd * ln1w[j] + ln1b[j]);
}

// ----------- DeepSets stage 3: in-place  e = LN(E + phi2_b)  (bf16 [1M,64])
__global__ __launch_bounds__(256) void ds_ln2_kernel(
    bf16_t* __restrict__ E, const float* __restrict__ phi2_b,
    const float* __restrict__ ln2w, const float* __restrict__ ln2b)
{
  const size_t i = (size_t)blockIdx.x * 256 + threadIdx.x;
  bf16_t* row = E + i * DH;
  float v[DH]; float sum = 0.f;
  #pragma unroll
  for (int j = 0; j < DH; ++j) { v[j] = (float)row[j] + phi2_b[j]; sum += v[j]; }
  const float mu = sum * (1.0f / DH);
  float sq = 0.f;
  #pragma unroll
  for (int j = 0; j < DH; ++j) { const float d = v[j] - mu; sq += d * d; }
  const float rstd = rsqrtf(sq * (1.0f / DH) + 1e-5f);
  #pragma unroll
  for (int j = 0; j < DH; ++j)
    row[j] = (bf16_t)((v[j] - mu) * rstd * ln2w[j] + ln2b[j]);
}

// ----------- ctx[t][j] = mean_k E[t,k,j]
__global__ __launch_bounds__(256) void ctx_kernel(
    const bf16_t* __restrict__ E, float* __restrict__ ctx)
{
  const int g = blockIdx.x * 256 + threadIdx.x;  // < T*DH
  const int t = g >> 6, j = g & 63;
  float s = 0.f;
  for (int k = 0; k < KSEL; ++k)
    s += (float)E[((size_t)t * KSEL + k) * DH + j];
  ctx[g] = s * (1.0f / KSEL);
}

// ----------- c = e + ctx  (in place, bf16)
__global__ __launch_bounds__(256) void addctx_kernel(
    bf16_t* __restrict__ E, const float* __restrict__ ctx)
{
  const size_t i = (size_t)blockIdx.x * 256 + threadIdx.x;
  const int t = (int)(i >> 8);
  const float* c = ctx + (size_t)t * DH;
  bf16_t* row = E + i * DH;
  #pragma unroll
  for (int j = 0; j < DH; ++j)
    row[j] = (bf16_t)((float)row[j] + c[j]);
}

// ----------- zero a_man (bf16 buffer, written as u32)
__global__ __launch_bounds__(256) void zero_u32_kernel(uint32_t* __restrict__ p, size_t n)
{
  const size_t i = (size_t)blockIdx.x * 256 + threadIdx.x;
  if (i < n) p[i] = 0u;
}

// ----------- scatter man back into d_ff lane layout (bf16)
__global__ __launch_bounds__(256) void scatter_kernel(
    const float* __restrict__ man, const int* __restrict__ topidx,
    bf16_t* __restrict__ aman)
{
  const size_t i = (size_t)blockIdx.x * 256 + threadIdx.x;
  const int t = (int)(i >> 8);
  const int idx = topidx[i];
  aman[(size_t)t * DF + idx] = (bf16_t)man[i];
}

// ----------- f32 -> bf16 weight conversion
__global__ __launch_bounds__(256) void cvt_kernel(
    const float* __restrict__ src, bf16_t* __restrict__ dst, size_t n)
{
  const size_t i = (size_t)blockIdx.x * 256 + threadIdx.x;
  if (i < n) dst[i] = (bf16_t)src[i];
}

// ---------------------------------------------------------------------------
extern "C" void kernel_launch(void* const* d_in, const int* in_sizes, int n_in,
                              void* d_out, int out_size, void* d_ws, size_t ws_size,
                              hipStream_t stream) {
  (void)in_sizes; (void)n_in; (void)out_size; (void)ws_size;
  const float* x       = (const float*)d_in[0];
  const float* W1      = (const float*)d_in[1];
  const float* W2      = (const float*)d_in[2];
  const float* Wr1     = (const float*)d_in[3];
  const float* Wr2     = (const float*)d_in[4];
  const float* ln_w    = (const float*)d_in[5];
  const float* ln_b    = (const float*)d_in[6];
  const float* phi1_w  = (const float*)d_in[7];
  const float* phi1_b  = (const float*)d_in[8];
  const float* pln1w   = (const float*)d_in[9];
  const float* pln1b   = (const float*)d_in[10];
  const float* phi2_w  = (const float*)d_in[11];
  const float* phi2_b  = (const float*)d_in[12];
  const float* pln2w   = (const float*)d_in[13];
  const float* pln2b   = (const float*)d_in[14];
  const float* rho1_w  = (const float*)d_in[15];
  const float* rho1_b  = (const float*)d_in[16];
  const float* rho2_w  = (const float*)d_in[17];
  const float* rho2_b  = (const float*)d_in[18];
  float* out = (float*)d_out;

  // ---- workspace carve (256B aligned); peak ~283 MB ----
  char* ws = (char*)d_ws;
  size_t off = 0;
  auto carve = [&](size_t bytes) -> char* {
    char* p = ws + off;
    off += (bytes + 255) & ~(size_t)255;
    return p;
  };
  const size_t TK = (size_t)T_TOK * KSEL;  // 1,048,576
  bf16_t* W1b   = (bf16_t*)carve((size_t)DF * DM * 2);
  bf16_t* Wr1b  = (bf16_t*)carve((size_t)DM * DM * 2);
  bf16_t* Wr2b  = (bf16_t*)carve((size_t)DF * DM * 2);
  bf16_t* W2b   = (bf16_t*)carve((size_t)DM * DF * 2);
  bf16_t* phi2b = (bf16_t*)carve((size_t)DH * DH2 * 2);
  bf16_t* rho1b = (bf16_t*)carve((size_t)(2 * DH) * DH * 2);
  bf16_t* xb    = (bf16_t*)carve((size_t)T_TOK * DM * 2);
  bf16_t* xnb   = (bf16_t*)carve((size_t)T_TOK * DM * 2);
  bf16_t* hb    = (bf16_t*)carve((size_t)T_TOK * DM * 2);
  // 128MB union: [scores(64MB) | z(64MB)] later reused as E (bf16 [T*K,64])
  char*   ubase  = carve((size_t)T_TOK * DF * 4 * 2);
  float*  scores = (float*)ubase;
  float*  z      = (float*)(ubase + (size_t)T_TOK * DF * 4);
  bf16_t* E      = (bf16_t*)ubase;            // alive only after scores/z dead
  int*    topidx = (int*)carve(TK * 4);
  bf16_t* A_ds   = (bf16_t*)carve(TK * DH2 * 2);
  float*  ctx    = (float*)carve((size_t)T_TOK * DH * 4);
  float*  man    = (float*)carve(TK * 4);
  bf16_t* aman   = (bf16_t*)carve((size_t)T_TOK * DF * 2);

  // ---- 1. weight conversions f32 -> bf16 ----
  cvt_kernel<<<(DF * DM) / 256, 256, 0, stream>>>(W1,  W1b,  (size_t)DF * DM);
  cvt_kernel<<<(DM * DM) / 256, 256, 0, stream>>>(Wr1, Wr1b, (size_t)DM * DM);
  cvt_kernel<<<(DF * DM) / 256, 256, 0, stream>>>(Wr2, Wr2b, (size_t)DF * DM);
  cvt_kernel<<<(DM * DF) / 256, 256, 0, stream>>>(W2,  W2b,  (size_t)DM * DF);
  cvt_kernel<<<(DH * DH2) / 256, 256, 0, stream>>>(phi2_w, phi2b, (size_t)DH * DH2);
  cvt_kernel<<<(2 * DH * DH) / 256, 256, 0, stream>>>(rho1_w, rho1b, (size_t)2 * DH * DH);

  // ---- 2. LN(x) and raw x -> bf16 ----
  lnx_kernel<<<T_TOK, 256, 0, stream>>>(x, ln_w, ln_b, xnb, xb);

  // ---- 3. router: h = gelu(x_norm @ Wr1^T)  [4096x1024] ----
  wmma_gemm_kernel<true, true><<<dim3(T_TOK / 64, DM / 128), 256, 0, stream>>>(
      xnb, Wr1b, hb, nullptr, T_TOK, DM, DM);

  // ---- 4. scores = h @ Wr2^T  [4096x4096] f32 ----
  wmma_gemm_kernel<false, false><<<dim3(T_TOK / 64, DF / 128), 256, 0, stream>>>(
      hb, Wr2b, scores, nullptr, T_TOK, DF, DM);

  // ---- 5. per-token top-256 ----
  topk_kernel<<<T_TOK, 256, 0, stream>>>(scores, topidx);

  // ---- 6. z = x @ W1^T  [4096x4096] f32 ----
  wmma_gemm_kernel<false, false><<<dim3(T_TOK / 64, DF / 128), 256, 0, stream>>>(
      xb, W1b, z, nullptr, T_TOK, DF, DM);

  // ---- 7. DeepSets phi1 + LN + gelu -> A_ds [1M x 32] bf16 ----
  ds_phi1_kernel<<<(unsigned)(TK / 256), 256, 0, stream>>>(
      z, topidx, phi1_w, phi1_b, pln1w, pln1b, A_ds);

  // ---- 8. E = A_ds @ phi2^T  [1M x 64] bf16 (overwrites scores/z region) ----
  wmma_gemm_kernel<false, true><<<dim3((unsigned)(TK / 64), 1), 256, 0, stream>>>(
      A_ds, phi2b, E, nullptr, (int)TK, DH, DH2);

  // ---- 9. e = LN(E + phi2_b) in place ----
  ds_ln2_kernel<<<(unsigned)(TK / 256), 256, 0, stream>>>(E, phi2_b, pln2w, pln2b);

  // ---- 10. ctx = mean_k e ;  c = e + ctx (in place) ----
  ctx_kernel<<<(T_TOK * DH) / 256, 256, 0, stream>>>(E, ctx);
  addctx_kernel<<<(unsigned)(TK / 256), 256, 0, stream>>>(E, ctx);

  // ---- 11. man = gelu(c @ rho1^T + b1) @ rho2^T + b2  (fused epilogue) ----
  wmma_rho_kernel<<<(unsigned)(TK / 64), 256, 0, stream>>>(
      E, rho1b, rho1_b, rho2_w, rho2_b, man, (int)TK);

  // ---- 12. scatter into a_man (zeroed) ----
  zero_u32_kernel<<<(unsigned)(((size_t)T_TOK * DF / 2) / 256), 256, 0, stream>>>(
      (uint32_t*)aman, (size_t)T_TOK * DF / 2);
  scatter_kernel<<<(unsigned)(TK / 256), 256, 0, stream>>>(man, topidx, aman);

  // ---- 13. out = a_man @ W2^T  [4096x1024] f32 ----
  wmma_gemm_kernel<false, false><<<dim3(T_TOK / 64, DM / 128), 256, 0, stream>>>(
      aman, W2b, out, nullptr, T_TOK, DM, DF);
}
